// ResonanceAttention_80126909874246
// MI455X (gfx1250) — matrix operand order repaired and checked by reading, not compiled, not run on hardware
//
#include <hip/hip_runtime.h>
#include <hip/hip_bf16.h>
#include <math.h>

// ---------------------------------------------------------------------------
// ResonanceAttention on MI455X (gfx1250, wave32, WMMA f16->f32)
// B=2, S=1024, D=1024, H=16, HD=64, 4 momentum softmax iterations.
// Phase modulation adds a per-head scalar to q and k; in softmax only the
// c_t * rowsum(k_j) term survives, so QK^T is computed ONCE.
// ---------------------------------------------------------------------------

#define BATCH 2
#define SEQ   1024
#define DMODEL 1024
#define NHEAD 16
#define HDIM  64

typedef _Float16 h8   __attribute__((ext_vector_type(8)));
typedef _Float16 v16h __attribute__((ext_vector_type(16)));
typedef float    v8f  __attribute__((ext_vector_type(8)));

static __device__ __forceinline__ v16h cat16(h8 lo, h8 hi) {
  return __builtin_shufflevector(lo, hi, 0,1,2,3,4,5,6,7,8,9,10,11,12,13,14,15);
}

// A fragment (16x32 f16, row-major source, leading dim ld):
// lane m = lane&15, g = lane>>4; k = g*8 + e (e<8) / 16 + g*8 + (e-8)
static __device__ __forceinline__ v16h load_a_frag(const _Float16* base, int ld,
                                                   int row0, int k0, int lane) {
  int m = lane & 15, g = lane >> 4;
  const _Float16* p = base + (size_t)(row0 + m) * ld + k0 + g * 8;
  h8 lo = *(const h8*)p;
  h8 hi = *(const h8*)(p + 16);
  return cat16(lo, hi);
}

// B fragment (32x16 f16) gathered from a row-major [N x K] matrix (B[k][n]=R[n][k]):
// lane n = lane&15, g = lane>>4; k = g*8 + e (e<8) / 16 + g*8 + (e-8)
static __device__ __forceinline__ v16h load_b_frag(const _Float16* base, int ld,
                                                   int n0, int k0, int lane) {
  int n = lane & 15, g = lane >> 4;
  const _Float16* p = base + (size_t)(n0 + n) * ld + k0 + g * 8;
  h8 lo = *(const h8*)p;
  h8 hi = *(const h8*)(p + 16);
  return cat16(lo, hi);
}

static __device__ __forceinline__ v8f wmma16(v16h a, v16h b, v8f c) {
  return __builtin_amdgcn_wmma_f32_16x16x32_f16(false, a, false, b, (short)0, c,
                                                false, false);
}

// ---------------------------------------------------------------------------
__global__ void k_f32_to_f16(const float* __restrict__ src,
                             _Float16* __restrict__ dst, int n) {
  int i = blockIdx.x * blockDim.x + threadIdx.x;
  if (i < n) dst[i] = (_Float16)src[i];
}

// per-row sum of K (f16) rows of length HDIM -> ksum (f32)
__global__ void k_rowsum(const _Float16* __restrict__ kh,
                         float* __restrict__ ksum, int nrows) {
  int r = blockIdx.x * blockDim.x + threadIdx.x;
  if (r >= nrows) return;
  const h8* p = (const h8*)(kh + (size_t)r * HDIM);
  float s = 0.f;
#pragma unroll
  for (int c = 0; c < HDIM / 8; ++c) {
    h8 v = p[c];
#pragma unroll
    for (int e = 0; e < 8; ++e) s += (float)v[e];
  }
  ksum[r] = s;
}

// ---------------------------------------------------------------------------
// GEMM: out[row, col] = sum_k A[row,k] * W[col,k] + bias[col]
// A: [2048 x 1024] f16 row-major, W: [1024 x 1024] f16 row-major.
// Each wave computes a 16x64 strip (4 N-tiles). Fragments for step k+32 are
// loaded BEFORE the 4 WMMAs of step k (double buffering), so the matrix pipe
// never waits on a just-issued load.
// mode 0: store f16 per-head layout  q[(b*H+h)*S + s][dl]
// mode 1: store f16 transposed-V     vT[(b*H+h)*HD + dl][s]
// mode 2: store f32 flat             out32[row*1024 + col]
// ---------------------------------------------------------------------------
__global__ __launch_bounds__(128) void k_gemm(const _Float16* __restrict__ A,
                                              const _Float16* __restrict__ W,
                                              const float* __restrict__ bias,
                                              _Float16* __restrict__ out16,
                                              float* __restrict__ out32,
                                              int mode) {
  int tid  = threadIdx.x;
  int lane = tid & 31, wid = tid >> 5;
  int tile_m = blockIdx.x * 4 + wid;   // 0..127
  int n_base = blockIdx.y * 64;        // 0..960 (all 4 waves share these cols)
  int row0 = tile_m * 16;

  v8f acc[4] = {{}, {}, {}, {}};

  // prologue: fragments for k = 0
  v16h a_cur = load_a_frag(A, DMODEL, row0, 0, lane);
  v16h b_cur[4];
#pragma unroll
  for (int j = 0; j < 4; ++j)
    b_cur[j] = load_b_frag(W, DMODEL, n_base + j * 16, 0, lane);

  for (int k0 = 0; k0 < DMODEL; k0 += 32) {
    int kn = (k0 + 32) & (DMODEL - 1);  // wraps to 0 on last iter (discarded)
    // issue next-step loads before consuming current fragments
    v16h a_nxt = load_a_frag(A, DMODEL, row0, kn, lane);
    v16h b_nxt[4];
#pragma unroll
    for (int j = 0; j < 4; ++j)
      b_nxt[j] = load_b_frag(W, DMODEL, n_base + j * 16, kn, lane);

    if (k0 + 512 < DMODEL) {
      // global_prefetch_b8: pull a future k-stripe of A rows and the W rows
      __builtin_prefetch(
          (const void*)(A + (size_t)(row0 + (lane & 15)) * DMODEL + k0 + 512),
          0, 1);
      __builtin_prefetch(
          (const void*)(W + (size_t)(n_base + lane * 2) * DMODEL + k0 + 512),
          0, 1);
    }

#pragma unroll
    for (int j = 0; j < 4; ++j) acc[j] = wmma16(a_cur, b_cur[j], acc[j]);

    a_cur = a_nxt;
#pragma unroll
    for (int j = 0; j < 4; ++j) b_cur[j] = b_nxt[j];
  }

  int n = lane & 15, g = lane >> 4;
#pragma unroll
  for (int j = 0; j < 4; ++j) {
    int col = n_base + j * 16 + n;
    float bval = bias[col];
    int h = col >> 6, dl = col & 63;
#pragma unroll
    for (int r = 0; r < 8; ++r) {
      int row = row0 + g * 8 + r;
      float v = acc[j][r] + bval;
      if (mode == 0) {
        int b = row >> 10, s = row & 1023;
        out16[(((size_t)(b * NHEAD + h)) * SEQ + s) * HDIM + dl] = (_Float16)v;
      } else if (mode == 1) {
        int b = row >> 10, s = row & 1023;
        out16[(((size_t)(b * NHEAD + h)) * HDIM + dl) * SEQ + s] = (_Float16)v;
      } else {
        out32[(size_t)row * DMODEL + col] = v;
      }
    }
  }
}

// ---------------------------------------------------------------------------
// Fused resonance attention. Grid: (B*H)*(S/16) blocks, 256 threads (8 waves).
// Each block handles one 16-row query block for one (b,h).
// Wave w owns score columns [w*128, w*128+128).
// ---------------------------------------------------------------------------
__global__ __launch_bounds__(256) void k_attn(const _Float16* __restrict__ qh,
                                              const _Float16* __restrict__ kh,
                                              const _Float16* __restrict__ vT,
                                              const float* __restrict__ ksum,
                                              const float* __restrict__ pf_in,
                                              _Float16* __restrict__ ctx) {
  __shared__ __align__(16) _Float16 sm_attn[8 * 16 * 128];  // 32 KB
  __shared__ float sm_red[8][16];
  __shared__ float sm_ctx[16 * 64];                          // 4 KB

  int tid = threadIdx.x, lane = tid & 31, wid = tid >> 5;
  int ln = lane & 15, g = lane >> 4;
  int bh = blockIdx.x >> 6, sblk = blockIdx.x & 63;
  int s0r = sblk * 16;

  const _Float16* qbase = qh + (size_t)bh * SEQ * HDIM;
  const _Float16* kbase = kh + (size_t)bh * SEQ * HDIM;

  // ---- S0 = q k^T for this wave's 8 column tiles (K = 64 -> 2 WMMAs each),
  //      with one-tile lookahead double buffering on the K fragments.
  v16h aq0 = load_a_frag(qbase, HDIM, s0r, 0, lane);
  v16h aq1 = load_a_frag(qbase, HDIM, s0r, 32, lane);

  v8f  s0[8];
  float ks[8];
  v16h b0c = load_b_frag(kbase, HDIM, (wid * 8) * 16, 0, lane);
  v16h b1c = load_b_frag(kbase, HDIM, (wid * 8) * 16, 32, lane);
#pragma unroll
  for (int t = 0; t < 8; ++t) {
    int j0  = (wid * 8 + t) * 16;
    int j0n = (wid * 8 + ((t + 1) & 7)) * 16;  // wraps (discarded) on last
    v16h b0n = load_b_frag(kbase, HDIM, j0n, 0, lane);
    v16h b1n = load_b_frag(kbase, HDIM, j0n, 32, lane);
    v8f c = {};
    c = wmma16(aq0, b0c, c);
    c = wmma16(aq1, b1c, c);
    s0[t] = c;
    ks[t] = ksum[(size_t)bh * SEQ + j0 + ln];
    b0c = b0n;
    b1c = b1n;
  }

  float pf = pf_in[bh & (NHEAD - 1)];

  float attn[8][8];
#pragma unroll
  for (int t = 0; t < 8; ++t)
#pragma unroll
    for (int r = 0; r < 8; ++r) attn[t][r] = 0.f;

  const float wmom[4] = {0.008f, 0.032f, 0.16f, 0.8f};  // momentum weights

#pragma unroll
  for (int it = 0; it < 4; ++it) {
    float phase = __sinf(1.57079632679f * (float)it);   // sin(2*pi*t/4)
    float cmod  = pf * phase;
    float bs    = (0.8f + 0.2f * (float)it) * 0.125f;   // beta_t / sqrt(HD)

    // ---- row max
    float pmax[8];
#pragma unroll
    for (int r = 0; r < 8; ++r) pmax[r] = -3.4e38f;
#pragma unroll
    for (int t = 0; t < 8; ++t)
#pragma unroll
      for (int r = 0; r < 8; ++r)
        pmax[r] = fmaxf(pmax[r], s0[t][r] + cmod * ks[t]);
#pragma unroll
    for (int r = 0; r < 8; ++r)
#pragma unroll
      for (int m = 1; m < 16; m <<= 1)
        pmax[r] = fmaxf(pmax[r], __shfl_xor(pmax[r], m, 32));
    __syncthreads();
    if (ln == 0) {
#pragma unroll
      for (int r = 0; r < 8; ++r) sm_red[wid][g * 8 + r] = pmax[r];
    }
    __syncthreads();
    float rmax[8];
#pragma unroll
    for (int r = 0; r < 8; ++r) {
      float m = sm_red[0][g * 8 + r];
#pragma unroll
      for (int w = 1; w < 8; ++w) m = fmaxf(m, sm_red[w][g * 8 + r]);
      rmax[r] = m;
    }

    // ---- row sum of exp
    float psum[8];
#pragma unroll
    for (int r = 0; r < 8; ++r) psum[r] = 0.f;
#pragma unroll
    for (int t = 0; t < 8; ++t)
#pragma unroll
      for (int r = 0; r < 8; ++r)
        psum[r] += __expf(bs * (s0[t][r] + cmod * ks[t] - rmax[r]));
#pragma unroll
    for (int r = 0; r < 8; ++r)
#pragma unroll
      for (int m = 1; m < 16; m <<= 1)
        psum[r] += __shfl_xor(psum[r], m, 32);
    __syncthreads();
    if (ln == 0) {
#pragma unroll
      for (int r = 0; r < 8; ++r) sm_red[wid][g * 8 + r] = psum[r];
    }
    __syncthreads();
    float coef[8];
#pragma unroll
    for (int r = 0; r < 8; ++r) {
      float s = 0.f;
#pragma unroll
      for (int w = 0; w < 8; ++w) s += sm_red[w][g * 8 + r];
      coef[r] = wmom[it] / s;
    }

    // ---- accumulate momentum-weighted softmax
#pragma unroll
    for (int t = 0; t < 8; ++t)
#pragma unroll
      for (int r = 0; r < 8; ++r)
        attn[t][r] += coef[r] * __expf(bs * (s0[t][r] + cmod * ks[t] - rmax[r]));
  }

  // ---- stage attn to LDS as f16 (C-layout -> memory), zero ctx accumulator
  _Float16* amine = sm_attn + (size_t)wid * 16 * 128;
#pragma unroll
  for (int t = 0; t < 8; ++t)
#pragma unroll
    for (int r = 0; r < 8; ++r)
      amine[(g * 8 + r) * 128 + t * 16 + ln] = (_Float16)attn[t][r];
  for (int i = tid; i < 16 * 64; i += 256) sm_ctx[i] = 0.f;
  __syncthreads();

  // ---- ctx partial = attn(16x128) @ v(128x64) via WMMA, A-frags from LDS
  v16h af[4];
#pragma unroll
  for (int kb = 0; kb < 4; ++kb)
    af[kb] = load_a_frag(amine, 128, 0, kb * 32, lane);

  const _Float16* vbase = vT + (size_t)bh * HDIM * SEQ;  // [64][1024] per head
#pragma unroll
  for (int dt = 0; dt < 4; ++dt) {
    // hoist all 4 B fragments of this strip ahead of the chained WMMAs
    v16h bf[4];
#pragma unroll
    for (int kb = 0; kb < 4; ++kb)
      bf[kb] = load_b_frag(vbase, SEQ, dt * 16, wid * 128 + kb * 32, lane);
    v8f cacc = {};
#pragma unroll
    for (int kb = 0; kb < 4; ++kb) cacc = wmma16(af[kb], bf[kb], cacc);
#pragma unroll
    for (int r = 0; r < 8; ++r)
      atomicAdd(&sm_ctx[(g * 8 + r) * 64 + dt * 16 + ln], cacc[r]);
  }
  __syncthreads();

  // ---- write context, interleaving heads back to [B, S, D] (f16)
  int b = bh >> 4, h = bh & 15;
  for (int i = tid; i < 16 * 64; i += 256) {
    int rr = i >> 6, dd = i & 63;
    ctx[((size_t)(b * SEQ + s0r + rr)) * DMODEL + h * HDIM + dd] =
        (_Float16)sm_ctx[i];
  }
}

// ---------------------------------------------------------------------------
extern "C" void kernel_launch(void* const* d_in, const int* in_sizes, int n_in,
                              void* d_out, int out_size, void* d_ws,
                              size_t ws_size, hipStream_t stream) {
  const float* x  = (const float*)d_in[0];
  const float* Wq = (const float*)d_in[1];
  const float* bq = (const float*)d_in[2];
  const float* Wk = (const float*)d_in[3];
  const float* bk = (const float*)d_in[4];
  const float* Wv = (const float*)d_in[5];
  const float* bv = (const float*)d_in[6];
  const float* Wo = (const float*)d_in[7];
  const float* bo = (const float*)d_in[8];
  // d_in[9] = resonance_bias: constant per head along the softmax axis -> cancels.
  const float* pf = (const float*)d_in[10];
  float* out = (float*)d_out;

  char* ws = (char*)d_ws;
  size_t off = 0;
  auto alloc = [&](size_t bytes) {
    char* p = ws + off;
    off += (bytes + 255) & ~(size_t)255;
    return p;
  };
  const size_t NTOK = (size_t)BATCH * SEQ;               // 2048
  _Float16* x_h  = (_Float16*)alloc(NTOK * DMODEL * 2);  // 4 MB
  _Float16* wq_h = (_Float16*)alloc((size_t)DMODEL * DMODEL * 2);
  _Float16* wk_h = (_Float16*)alloc((size_t)DMODEL * DMODEL * 2);
  _Float16* wv_h = (_Float16*)alloc((size_t)DMODEL * DMODEL * 2);
  _Float16* wo_h = (_Float16*)alloc((size_t)DMODEL * DMODEL * 2);
  _Float16* q_h  = (_Float16*)alloc(NTOK * DMODEL * 2);  // [B*H][S][64]
  _Float16* k_h  = (_Float16*)alloc(NTOK * DMODEL * 2);  // [B*H][S][64]
  _Float16* vT_h = (_Float16*)alloc(NTOK * DMODEL * 2);  // [B*H][64][S]
  float*    ksmb = (float*)alloc((size_t)BATCH * NHEAD * SEQ * 4);
  _Float16* ctx_h = (_Float16*)alloc(NTOK * DMODEL * 2);

  const int NX = BATCH * SEQ * DMODEL;   // 2097152
  const int NW = DMODEL * DMODEL;        // 1048576
  k_f32_to_f16<<<(NX + 255) / 256, 256, 0, stream>>>(x, x_h, NX);
  k_f32_to_f16<<<(NW + 255) / 256, 256, 0, stream>>>(Wq, wq_h, NW);
  k_f32_to_f16<<<(NW + 255) / 256, 256, 0, stream>>>(Wk, wk_h, NW);
  k_f32_to_f16<<<(NW + 255) / 256, 256, 0, stream>>>(Wv, wv_h, NW);
  k_f32_to_f16<<<(NW + 255) / 256, 256, 0, stream>>>(Wo, wo_h, NW);

  dim3 gg(32, 16);  // (M/16/4, N/64)
  k_gemm<<<gg, 128, 0, stream>>>(x_h, wq_h, bq, q_h,  nullptr, 0);
  k_gemm<<<gg, 128, 0, stream>>>(x_h, wk_h, bk, k_h,  nullptr, 0);
  k_gemm<<<gg, 128, 0, stream>>>(x_h, wv_h, bv, vT_h, nullptr, 1);

  const int NROWS = BATCH * NHEAD * SEQ;  // 32768
  k_rowsum<<<(NROWS + 255) / 256, 256, 0, stream>>>(k_h, ksmb, NROWS);

  k_attn<<<BATCH * NHEAD * (SEQ / 16), 256, 0, stream>>>(q_h, k_h, vT_h, ksmb,
                                                         pf, ctx_h);

  k_gemm<<<gg, 128, 0, stream>>>(ctx_h, wo_h, bo, nullptr, out, 2);
}
